// SASRecMoEHard_69930657513760
// MI455X (gfx1250) — matrix operand order repaired and checked by reading, not compiled, not run on hardware
//
#include <hip/hip_runtime.h>

typedef _Float16 half_t;
typedef __attribute__((ext_vector_type(16))) _Float16 v16h;
typedef __attribute__((ext_vector_type(8)))  _Float16 v8h;
typedef __attribute__((ext_vector_type(8)))  float    v8f;

#define NB   128     // batch
#define NT   200     // seq len
#define ND   256     // model dim
#define NH   4       // heads
#define NHD  64      // head dim
#define NE   4       // experts
#define NL   2       // layers
#define TPAD 224     // padded T for vT buffer (multiple of 32)
#define QTILES 13    // ceil(200/16)
#define ROWS (NB*NT) // 25600

#define WMMA_F16(A,B,C) __builtin_amdgcn_wmma_f32_16x16x32_f16(false,(A),false,(B),(short)0,(C),false,false)

static __device__ inline v8f zero8() {
    v8f z;
#pragma unroll
    for (int i = 0; i < 8; ++i) z[i] = 0.0f;
    return z;
}

static __device__ inline v16h mk16(v8h lo, v8h hi) {
    v16h r;
#pragma unroll
    for (int i = 0; i < 8; ++i) { r[i] = lo[i]; r[i + 8] = hi[i]; }
    return r;
}

// A-fragment: elements 0..7 at p, elements 8..15 at p+16 (K offset +16)
static __device__ inline v16h ldA(const half_t* p) {
    return mk16(*(const v8h*)p, *(const v8h*)(p + 16));
}
// B-fragment: elements 0..7 at p, elements 8..15 at p+16 (K offset +16)
static __device__ inline v16h ldB(const half_t* p) {
    return mk16(*(const v8h*)p, *(const v8h*)(p + 16));
}

// ---------------------------------------------------------------- convert f32 -> f16
__global__ __launch_bounds__(256) void cvt_kernel(const float* __restrict__ x,
                                                  half_t* __restrict__ y, int n) {
    int i = blockIdx.x * 256 + threadIdx.x;
    if (i < n) y[i] = (half_t)x[i];
}

// ---------------------------------------------------------------- embedding
__global__ __launch_bounds__(256) void embed_kernel(const int* __restrict__ logs,
                                                    const float* __restrict__ item,
                                                    const float* __restrict__ pose,
                                                    float* __restrict__ sF,
                                                    half_t* __restrict__ sH) {
    int row = blockIdx.x;              // b*NT + t
    int t   = row % NT;
    int tid = threadIdx.x;
    int id  = logs[row];
    int pos = (id != 0) ? (t + 1) : 0;
    float v = item[(size_t)id * ND + tid] * 16.0f + pose[(size_t)pos * ND + tid];
    size_t idx = (size_t)row * ND + tid;
    sF[idx] = v;
    sH[idx] = (half_t)v;
}

// ---------------------------------------------------------------- layernorm (row = 256 elems)
__global__ __launch_bounds__(256) void ln_kernel(const float* __restrict__ x,
                                                 const float* __restrict__ g,
                                                 const float* __restrict__ be,
                                                 float* __restrict__ yF,
                                                 half_t* __restrict__ yH) {
    __shared__ float red[8];
    int row = blockIdx.x, tid = threadIdx.x;
    size_t base = (size_t)row * ND;
    float v = x[base + tid];
    float s = v;
#pragma unroll
    for (int m = 1; m < 32; m <<= 1) s += __shfl_xor(s, m, 32);
    if ((tid & 31) == 0) red[tid >> 5] = s;
    __syncthreads();
    float tot = 0.0f;
#pragma unroll
    for (int i = 0; i < 8; ++i) tot += red[i];
    float mean = tot * (1.0f / 256.0f);
    __syncthreads();
    float d = v - mean;
    float sq = d * d;
#pragma unroll
    for (int m = 1; m < 32; m <<= 1) sq += __shfl_xor(sq, m, 32);
    if ((tid & 31) == 0) red[tid >> 5] = sq;
    __syncthreads();
    float vt = 0.0f;
#pragma unroll
    for (int i = 0; i < 8; ++i) vt += red[i];
    float var = vt * (1.0f / 256.0f);
    float y = d / sqrtf(var + 1e-8f) * g[tid] + be[tid];
    if (yF) yF[base + tid] = y;
    if (yH) yH[base + tid] = (half_t)y;
}

// ---------------------------------------------------------------- QKV projection (WMMA)
// Each wave computes a 16x64 strip (4 N-tiles), A fragment reused across tiles.
// 12 column-groups of 64: groups 0-3 = q (X=qn), 4-7 = k, 8-11 = v.
// q,k -> qkH (ROWS x 512 row-major);  v -> vTH transposed (b,h,hd,TPAD).
__global__ __launch_bounds__(128) void qkv_kernel(const half_t* __restrict__ qnH,
                                                  const half_t* __restrict__ seqsH,
                                                  const half_t* __restrict__ Wl,   // layer base (E,768,256)
                                                  const float* __restrict__ Bl,    // layer base (E,768)
                                                  const int* __restrict__ eids,
                                                  half_t* __restrict__ qkH,
                                                  half_t* __restrict__ vTH) {
    int wid  = threadIdx.x >> 5;
    int lane = threadIdx.x & 31;
    int tile = blockIdx.x * 4 + wid;          // NB*QTILES*12 tiles total
    int ng = tile % 12; int tmp = tile / 12;
    int tt = tmp % QTILES; int b = tmp / QTILES;
    int li = lane & 15, hig = lane >> 4;
    int n0g = ng * 64, t0 = tt * 16;
    int e = eids[b];
    const half_t* W  = Wl + (size_t)e * 768 * ND;
    const float*  bp = Bl + (size_t)e * 768;
    const half_t* X  = (n0g < 256) ? qnH : seqsH;

    int trow = t0 + li; trow = trow < (NT - 1) ? trow : (NT - 1);
    const half_t* xrow = X + ((size_t)b * NT + trow) * ND;
    const half_t* wrow = W + (size_t)(n0g + li) * ND;

    v8f acc[4];
#pragma unroll
    for (int nn = 0; nn < 4; ++nn) acc[nn] = zero8();
#pragma unroll
    for (int kk = 0; kk < 8; ++kk) {
        int kb = kk * 32;
        v16h A = ldA(xrow + kb + 8 * hig);
#pragma unroll
        for (int nn = 0; nn < 4; ++nn) {
            v16h Bf = ldB(wrow + (size_t)nn * 16 * ND + kb + 8 * hig);
            acc[nn] = WMMA_F16(A, Bf, acc[nn]);
        }
    }
    float bv[4];
#pragma unroll
    for (int nn = 0; nn < 4; ++nn) bv[nn] = bp[n0g + nn * 16 + li];

    if (n0g < 512) {
#pragma unroll
        for (int nn = 0; nn < 4; ++nn)
#pragma unroll
            for (int r = 0; r < 8; ++r) {
                int t = t0 + r + 8 * hig;
                if (t < NT)
                    qkH[((size_t)b * NT + t) * 512 + n0g + nn * 16 + li] = (half_t)(acc[nn][r] + bv[nn]);
            }
    } else {
#pragma unroll
        for (int nn = 0; nn < 4; ++nn) {
            int f = n0g + nn * 16 + li - 512;
            int hh = f >> 6, hd = f & 63;
            half_t* col = vTH + (((size_t)b * NH + hh) * NHD + hd) * TPAD;
#pragma unroll
            for (int r = 0; r < 8; ++r) {
                int t = t0 + r + 8 * hig;
                if (t < NT) col[t] = (half_t)(acc[nn][r] + bv[nn]);
            }
        }
    }
}

// ---------------------------------------------------------------- flash attention (one wave per (b,h,qtile))
__global__ __launch_bounds__(32) void attn_kernel(const half_t* __restrict__ qk,
                                                  const half_t* __restrict__ vT,
                                                  half_t* __restrict__ o) {
    __shared__ __align__(16) half_t P[16 * 32];
    int blk = blockIdx.x;
    int qt = blk % QTILES; int tmp = blk / QTILES;
    int h = tmp & 3; int b = tmp >> 2;
    int lane = threadIdx.x, li = lane & 15, hig = lane >> 4;
    int q0 = qt * 16;

    float m[8], l[8];
    v8f accO[4];
#pragma unroll
    for (int r = 0; r < 8; ++r) { m[r] = -1e30f; l[r] = 0.0f; }
#pragma unroll
    for (int nn = 0; nn < 4; ++nn) accO[nn] = zero8();

    int qrow = q0 + li; qrow = qrow < (NT - 1) ? qrow : (NT - 1);
    const half_t* qbase = qk + ((size_t)b * NT + qrow) * 512 + h * NHD;
    v16h Aq[2];
#pragma unroll
    for (int dk = 0; dk < 2; ++dk) Aq[dk] = ldA(qbase + dk * 32 + 8 * hig);

    int kend = q0 + 15; if (kend > NT - 1) kend = NT - 1;
    for (int k0 = 0; k0 <= kend; k0 += 32) {
        v8f s0 = zero8(), s1 = zero8();
#pragma unroll
        for (int cj = 0; cj < 2; ++cj) {
            int kr = k0 + cj * 16 + li; kr = kr < (NT - 1) ? kr : (NT - 1);
            const half_t* kbase = qk + ((size_t)b * NT + kr) * 512 + 256 + h * NHD;
#pragma unroll
            for (int dk = 0; dk < 2; ++dk) {
                v16h Bf = ldB(kbase + dk * 32 + 8 * hig);
                if (cj == 0) s0 = WMMA_F16(Aq[dk], Bf, s0);
                else         s1 = WMMA_F16(Aq[dk], Bf, s1);
            }
        }
        float scl[8];
#pragma unroll
        for (int r = 0; r < 8; ++r) {
            int qq = q0 + r + 8 * hig;
            int kc0 = k0 + li, kc1 = k0 + 16 + li;
            float a0 = s0[r] * 0.125f, a1 = s1[r] * 0.125f;
            a0 = (kc0 > qq || kc0 >= NT) ? -1e30f : a0;
            a1 = (kc1 > qq || kc1 >= NT) ? -1e30f : a1;
            s0[r] = a0; s1[r] = a1;
            float mx = fmaxf(a0, a1);
#pragma unroll
            for (int msk = 1; msk < 16; msk <<= 1) mx = fmaxf(mx, __shfl_xor(mx, msk, 32));
            float mn = fmaxf(m[r], mx);
            scl[r] = __expf(m[r] - mn);
            m[r] = mn;
        }
        __syncthreads();
#pragma unroll
        for (int r = 0; r < 8; ++r) {
            float p0 = __expf(s0[r] - m[r]);
            float p1 = __expf(s1[r] - m[r]);
            P[(r + 8 * hig) * 32 + li]      = (half_t)p0;
            P[(r + 8 * hig) * 32 + 16 + li] = (half_t)p1;
            float su = p0 + p1;
#pragma unroll
            for (int msk = 1; msk < 16; msk <<= 1) su += __shfl_xor(su, msk, 32);
            l[r] = l[r] * scl[r] + su;
#pragma unroll
            for (int nn = 0; nn < 4; ++nn) accO[nn][r] *= scl[r];
        }
        __syncthreads();
        v16h Ap = ldA(&P[li * 32 + 8 * hig]);
#pragma unroll
        for (int nn = 0; nn < 4; ++nn) {
            const half_t* vb = vT + (((size_t)b * NH + h) * NHD + nn * 16 + li) * TPAD + k0 + 8 * hig;
            v16h Bf = ldB(vb);
            accO[nn] = WMMA_F16(Ap, Bf, accO[nn]);
        }
        __syncthreads();
    }
#pragma unroll
    for (int nn = 0; nn < 4; ++nn)
#pragma unroll
        for (int r = 0; r < 8; ++r) {
            int t = q0 + r + 8 * hig;
            if (t < NT)
                o[((size_t)b * NT + t) * ND + h * NHD + nn * 16 + li] = (half_t)(accO[nn][r] / l[r]);
        }
}

// ---------------------------------------------------------------- generic 256-col GEMM: Y = X@W^T + bias (+resid, relu)
// Each wave computes a 16x64 strip (4 N-tiles); A fragment reused 4x per k-step.
__global__ __launch_bounds__(128) void gemm256_kernel(const half_t* __restrict__ X,
                                                      const half_t* __restrict__ Wl,
                                                      size_t wStride,
                                                      const int* __restrict__ eids,
                                                      const float* __restrict__ Bl,
                                                      int bStride,
                                                      const float* __restrict__ resid,
                                                      int relu,
                                                      float* __restrict__ outF,
                                                      half_t* __restrict__ outH) {
    int wid  = threadIdx.x >> 5;
    int lane = threadIdx.x & 31;
    int tile = blockIdx.x * 4 + wid;          // NB*QTILES*4 tiles total
    int ng = tile & 3; int tmp = tile >> 2;
    int tt = tmp % QTILES; int b = tmp / QTILES;
    int li = lane & 15, hig = lane >> 4;
    int n0g = ng * 64, t0 = tt * 16;

    const half_t* W = Wl;
    const float* bp = Bl;
    if (eids) {
        int e = eids[b];
        W += (size_t)e * wStride;
        bp += (size_t)e * bStride;
    }
    int trow = t0 + li; trow = trow < (NT - 1) ? trow : (NT - 1);
    const half_t* xrow = X + ((size_t)b * NT + trow) * ND;
    const half_t* wrow = W + (size_t)(n0g + li) * ND;

    v8f acc[4];
#pragma unroll
    for (int nn = 0; nn < 4; ++nn) acc[nn] = zero8();
#pragma unroll
    for (int kk = 0; kk < 8; ++kk) {
        int kb = kk * 32;
        v16h A = ldA(xrow + kb + 8 * hig);
#pragma unroll
        for (int nn = 0; nn < 4; ++nn) {
            v16h Bf = ldB(wrow + (size_t)nn * 16 * ND + kb + 8 * hig);
            acc[nn] = WMMA_F16(A, Bf, acc[nn]);
        }
    }
#pragma unroll
    for (int nn = 0; nn < 4; ++nn) {
        float bv = bp[n0g + nn * 16 + li];
#pragma unroll
        for (int r = 0; r < 8; ++r) {
            int t = t0 + r + 8 * hig;
            if (t < NT) {
                size_t idx = ((size_t)b * NT + t) * ND + n0g + nn * 16 + li;
                float v = acc[nn][r] + bv;
                if (resid) v += resid[idx];
                if (relu)  v = fmaxf(v, 0.0f);
                if (outF) outF[idx] = v;
                if (outH) outH[idx] = (half_t)v;
            }
        }
    }
}

// ---------------------------------------------------------------- final logits
__global__ __launch_bounds__(32) void logits_kernel(const float* __restrict__ feats,
                                                    const float* __restrict__ item,
                                                    const int* __restrict__ pos,
                                                    const int* __restrict__ neg,
                                                    float* __restrict__ out) {
    int row = blockIdx.x;
    int lane = threadIdx.x;
    const float* f = feats + (size_t)row * ND;
    const float* ep = item + (size_t)pos[row] * ND;
    const float* en = item + (size_t)neg[row] * ND;
    float sp = 0.0f, sn = 0.0f;
#pragma unroll
    for (int i = 0; i < 8; ++i) {
        int c = lane + 32 * i;
        float fv = f[c];
        sp += fv * ep[c];
        sn += fv * en[c];
    }
#pragma unroll
    for (int m = 1; m < 32; m <<= 1) {
        sp += __shfl_xor(sp, m, 32);
        sn += __shfl_xor(sn, m, 32);
    }
    if (lane == 0) {
        out[row] = sp;
        out[ROWS + row] = sn;
    }
}

// ================================================================ host
extern "C" void kernel_launch(void* const* d_in, const int* in_sizes, int n_in,
                              void* d_out, int out_size, void* d_ws, size_t ws_size,
                              hipStream_t stream) {
    (void)in_sizes; (void)n_in; (void)out_size; (void)ws_size;
    const int*   log_seqs  = (const int*)d_in[0];
    const int*   pos_seqs  = (const int*)d_in[1];
    const int*   neg_seqs  = (const int*)d_in[2];
    const int*   expert_ids= (const int*)d_in[3];
    const float* item_emb  = (const float*)d_in[4];
    const float* pos_emb   = (const float*)d_in[5];
    const float* attn_ln_s = (const float*)d_in[6];
    const float* attn_ln_b = (const float*)d_in[7];
    const float* in_w      = (const float*)d_in[8];
    const float* in_b      = (const float*)d_in[9];
    const float* out_w     = (const float*)d_in[10];
    const float* out_b     = (const float*)d_in[11];
    const float* fwd_ln_s  = (const float*)d_in[12];
    const float* fwd_ln_b  = (const float*)d_in[13];
    const float* ffn_w1    = (const float*)d_in[14];
    const float* ffn_b1    = (const float*)d_in[15];
    const float* ffn_w2    = (const float*)d_in[16];
    const float* ffn_b2    = (const float*)d_in[17];
    const float* last_ln_s = (const float*)d_in[18];
    const float* last_ln_b = (const float*)d_in[19];
    float* out = (float*)d_out;

    // workspace carving (d_ws is 256B aligned; keep every chunk 256B aligned)
    char* p = (char*)d_ws;
    auto carve = [&](size_t bytes) -> char* {
        char* r = p;
        p += (bytes + 255) & ~(size_t)255;
        return r;
    };
    float*  seqsF = (float*)carve((size_t)ROWS * ND * 4);
    float*  qnF   = (float*)carve((size_t)ROWS * ND * 4);
    float*  s2F   = (float*)carve((size_t)ROWS * ND * 4);
    half_t* seqsH = (half_t*)carve((size_t)ROWS * ND * 2);
    half_t* qnH   = (half_t*)carve((size_t)ROWS * ND * 2);
    half_t* qkH   = (half_t*)carve((size_t)ROWS * 512 * 2);
    half_t* vTH   = (half_t*)carve((size_t)NB * NH * NHD * TPAD * 2);
    half_t* oH    = (half_t*)carve((size_t)ROWS * ND * 2);
    half_t* hH    = (half_t*)carve((size_t)ROWS * ND * 2);
    half_t* inwH  = (half_t*)carve((size_t)NL * NE * 768 * ND * 2);
    half_t* outwH = (half_t*)carve((size_t)NL * NE * ND * ND * 2);
    half_t* w1H   = (half_t*)carve((size_t)NL * ND * ND * 2);
    half_t* w2H   = (half_t*)carve((size_t)NL * ND * ND * 2);

    // weight conversion
    {
        int n1 = NL * NE * 768 * ND;
        int n2 = NL * NE * ND * ND;
        int n3 = NL * ND * ND;
        cvt_kernel<<<(n1 + 255) / 256, 256, 0, stream>>>(in_w,   inwH,  n1);
        cvt_kernel<<<(n2 + 255) / 256, 256, 0, stream>>>(out_w,  outwH, n2);
        cvt_kernel<<<(n3 + 255) / 256, 256, 0, stream>>>(ffn_w1, w1H,   n3);
        cvt_kernel<<<(n3 + 255) / 256, 256, 0, stream>>>(ffn_w2, w2H,   n3);
    }

    embed_kernel<<<ROWS, 256, 0, stream>>>(log_seqs, item_emb, pos_emb, seqsF, seqsH);

    const int gemmBlocks = NB * QTILES * 4 / 4;   // 1664 blocks of 4 waves
    const int qkvBlocks  = NB * QTILES * 12 / 4;  // 4992 blocks of 4 waves

    for (int l = 0; l < NL; ++l) {
        // qn = LN(seqs)
        ln_kernel<<<ROWS, 256, 0, stream>>>(seqsF, attn_ln_s + l * ND, attn_ln_b + l * ND, qnF, qnH);
        // q,k,v projections (expert-selected)
        qkv_kernel<<<qkvBlocks, 128, 0, stream>>>(
            qnH, seqsH,
            inwH + (size_t)l * NE * 768 * ND,
            in_b + (size_t)l * NE * 768,
            expert_ids, qkH, vTH);
        // attention
        attn_kernel<<<NB * NH * QTILES, 32, 0, stream>>>(qkH, vTH, oH);
        // out projection + residual(qn) -> s2
        gemm256_kernel<<<gemmBlocks, 128, 0, stream>>>(
            oH, outwH + (size_t)l * NE * ND * ND, (size_t)ND * ND, expert_ids,
            out_b + (size_t)l * NE * ND, ND,
            qnF, 0, s2F, (half_t*)nullptr);
        // seqs = LN(s2)
        ln_kernel<<<ROWS, 256, 0, stream>>>(s2F, fwd_ln_s + l * ND, fwd_ln_b + l * ND, seqsF, seqsH);
        // h = relu(seqs @ w1^T + b1)
        gemm256_kernel<<<gemmBlocks, 128, 0, stream>>>(
            seqsH, w1H + (size_t)l * ND * ND, 0, (const int*)nullptr,
            ffn_b1 + (size_t)l * ND, 0,
            (const float*)nullptr, 1, (float*)nullptr, hH);
        // seqs = seqs + h @ w2^T + b2   (in-place residual; writes f32 + f16)
        gemm256_kernel<<<gemmBlocks, 128, 0, stream>>>(
            hH, w2H + (size_t)l * ND * ND, 0, (const int*)nullptr,
            ffn_b2 + (size_t)l * ND, 0,
            seqsF, 0, seqsF, seqsH);
    }

    // feats = LN(seqs, last)
    ln_kernel<<<ROWS, 256, 0, stream>>>(seqsF, last_ln_s, last_ln_b, qnF, (half_t*)nullptr);
    // logits
    logits_kernel<<<ROWS, 32, 0, stream>>>(qnF, item_emb, pos_seqs, neg_seqs, out);
}
